// GAT_9955734192547
// MI455X (gfx1250) — matrix operand order repaired and checked by reading, not compiled, not run on hardware
//
#include <hip/hip_runtime.h>
#include <hip/hip_bf16.h>
#include <stdint.h>

#define ALPHA_SLOPE 0.2f
#define MASK_VAL    (-9.0e15f)
#define BB 8
#define NN 2048
#define FF 256

typedef __attribute__((ext_vector_type(16))) _Float16 v16h;
typedef __attribute__((ext_vector_type(8)))  float    v8f;

union Frag16 { v16h v; uint32_t u[8]; _Float16 h[16]; };

#if defined(__gfx1250__) && __has_builtin(__builtin_amdgcn_global_load_async_to_lds_b128) && __has_builtin(__builtin_amdgcn_s_wait_asynccnt)
#define ASYNC_LDS 1
#else
#define ASYNC_LDS 0
#endif

#if ASYNC_LDS
// The builtin's param 1 is 'vector_size(16) int __device__ *' (AS1, global),
// param 2 is the LDS-side pointer. Match the pointee type exactly.
typedef int v4i_gcc __attribute__((vector_size(16)));
typedef __attribute__((address_space(1))) v4i_gcc g_v4i;
typedef __attribute__((address_space(3))) v4i_gcc l_v4i;

__device__ __forceinline__ void async_b128_to_lds(const void* gsrc, void* ldst) {
  // AS1 from full 64-bit VA; AS3 = low 32 bits of generic LDS address (aperture rule).
  // uintptr_t round-trip strips const and performs int->ptr (always legal).
  g_v4i* gp = (g_v4i*)(uintptr_t)gsrc;
  l_v4i* lp = (l_v4i*)(uint32_t)(uintptr_t)ldst;
  __builtin_amdgcn_global_load_async_to_lds_b128(gp, lp, 0, 0);
}
#endif

// A-matrix (16x32 f16) dword d of lane-group `half` holds K = kkA, kkA+1
__device__ __forceinline__ int kkA(int d, int half) {
  return (d < 4 ? 2 * d : 16 + 2 * (d - 4)) + 8 * half;
}
// B-matrix (32x16 f16): same K order as A (dword d of lane-group `half` holds K = kkA(d, half), +1)
__device__ __forceinline__ int kkB(int d, int half) {
  return (d < 4 ? 2 * d : 16 + 2 * (d - 4)) + 8 * half;
}

__device__ __forceinline__ float lrelu(float v) {
  return v > 0.0f ? v : ALPHA_SLOPE * v;
}

// ---------------- fp32 -> f16 convert ----------------
__global__ void __launch_bounds__(256)
convert_x_kernel(const float* __restrict__ x, _Float16* __restrict__ h16, int n) {
  int i = blockIdx.x * 256 + threadIdx.x;
  if (i < n) h16[i] = (_Float16)x[i];
}

// W [f][o] fp32 -> w16t [o][f] f16 (transposed so B-operand reads K-contiguous)
__global__ void __launch_bounds__(256)
convert_w_kernel(const float* __restrict__ W, _Float16* __restrict__ w16t) {
  int f = blockIdx.x;      // 0..255
  int o = threadIdx.x;     // 0..255
  w16t[o * FF + f] = (_Float16)W[f * FF + o];
}

// ---------------- Wh = h @ W  (WMMA f16->f32), writes Wh transposed ----------------
__global__ void __launch_bounds__(128)
gemm_wh_kernel(const _Float16* __restrict__ h16,
               const _Float16* __restrict__ w16t,
               _Float16* __restrict__ wh16t) {
  const int lane = threadIdx.x & 31;
  const int wave = threadIdx.x >> 5;
  const int m0   = blockIdx.x * 16;            // flat node tile (b*N + n0)
  const int ot   = blockIdx.y * 4 + wave;      // o-tile 0..15
  const int o0   = ot * 16;
  const int half = lane >> 4;
  const int lid  = lane & 15;

  const uint32_t* arow = (const uint32_t*)(h16  + (size_t)(m0 + lid) * FF);
  const uint32_t* brow = (const uint32_t*)(w16t + (size_t)(o0 + lid) * FF);

  v8f acc = {};
  for (int kb = 0; kb < FF; kb += 32) {
    Frag16 af, bf;
#pragma unroll
    for (int d = 0; d < 8; ++d) {
      af.u[d] = arow[(kb + kkA(d, half)) >> 1];
      bf.u[d] = brow[(kb + kkB(d, half)) >> 1];
    }
    acc = __builtin_amdgcn_wmma_f32_16x16x32_f16(false, af.v, false, bf.v,
                                                 (short)0, acc, false, false);
  }

  const int bidx = m0 / NN;
  const int nloc = (m0 % NN) + 8 * half;
  _Float16 tmp[8];
#pragma unroll
  for (int r = 0; r < 8; ++r) tmp[r] = (_Float16)acc[r];
  *(uint4*)(wh16t + ((size_t)(bidx * FF + o0 + lid) * NN + nloc)) = *(const uint4*)tmp;
}

// ---------------- f1/f2 = Wh @ a1, Wh @ a2 ----------------
__global__ void __launch_bounds__(256)
fvec_kernel(const _Float16* __restrict__ wh16t, const float* __restrict__ Avec,
            float* __restrict__ f1, float* __restrict__ f2) {
  const int b = blockIdx.y;
  const int n = blockIdx.x * 256 + threadIdx.x;
  float s1 = 0.0f, s2 = 0.0f;
  for (int o = 0; o < FF; ++o) {
    float w = (float)wh16t[((size_t)(b * FF + o)) * NN + n];   // coalesced over n
    s1 = fmaf(w, Avec[o], s1);
    s2 = fmaf(w, Avec[FF + o], s2);
  }
  f1[b * NN + n] = s1;
  f2[b * NN + n] = s2;
}

// ---------------- per-column softmax stats (axis=1): online max + sum ----------------
__global__ void __launch_bounds__(256)
colstats_kernel(const int* __restrict__ adj, const float* __restrict__ f1,
                const float* __restrict__ f2, float* __restrict__ Mb,
                float* __restrict__ invS) {
  const int b = blockIdx.y;
  const int j = blockIdx.x * 256 + threadIdx.x;
  const float f2j = f2[b * NN + j];
  const float* f1b = f1 + b * NN;
  const int* acol = adj + (size_t)b * NN * NN + j;

  float m = MASK_VAL, s = 0.0f;
  for (int i = 0; i < NN; ++i) {
    if ((i & 63) == 0 && i + 64 < NN)
      __builtin_prefetch(acol + (size_t)(i + 64) * NN, 0, 3);
    int av = acol[(size_t)i * NN];
    float v = (av > 0) ? lrelu(f1b[i] + f2j) : MASK_VAL;
    if (v > m) { s = s * __expf(m - v) + 1.0f; m = v; }
    else       { s += __expf(v - m); }
  }
  Mb[b * NN + j]   = m;
  invS[b * NN + j] = 1.0f / s;
}

// ---------------- fused attention GEMM: out[i,o] = sum_j p[i,j] * Wh[j,o] ----------------
// Block: 256 threads (8 waves). i-tile = 32 rows, o = 256 cols (32 per wave).
// adj tile (32x32 int) double-buffered in LDS via async-to-LDS b128; P tile 32x32 f16 in
// LDS feeds two A-fragments per wave; 4 WMMAs per wave per k-step.
__global__ void __launch_bounds__(256)
attn_kernel(const int* __restrict__ adj, const float* __restrict__ f1,
            const float* __restrict__ f2, const float* __restrict__ Mb,
            const float* __restrict__ invS, const _Float16* __restrict__ wh16t,
            _Float16* __restrict__ h16_out, float* __restrict__ f32_out,
            int write_f32) {
  __shared__ __align__(16) _Float16 Pt[32 * 32];
#if ASYNC_LDS
  __shared__ __align__(16) int AdjT[2][32 * 32];
#endif

  const int tid  = threadIdx.x;
  const int lane = tid & 31;
  const int wave = tid >> 5;
  const int gi0  = blockIdx.x * 32;     // flat b*N + i0
  const int b    = gi0 / NN;
  const int half = lane >> 4;
  const int lid  = lane & 15;
  const int o0   = wave * 32 + lid;     // c*0 column; c*1 column = o0 + 16

  // producer role: thread computes p for row ii, j-quad jq..jq+3
  const int ii = tid >> 3;              // 0..31
  const int jq = (tid & 7) * 4;         // 0,4,...,28
  const float f1v = f1[gi0 + ii];
  const int* arow = adj + (size_t)(gi0 + ii) * NN;
  const float* f2b = f2 + b * NN;
  const float* Mbb = Mb + b * NN;
  const float* iSb = invS + b * NN;

  const uint32_t* brow0 = (const uint32_t*)(wh16t + ((size_t)(b * FF + o0))      * NN);
  const uint32_t* brow1 = (const uint32_t*)(wh16t + ((size_t)(b * FF + o0 + 16)) * NN);

  v8f c00 = {}, c01 = {}, c10 = {}, c11 = {};

#if ASYNC_LDS
  // prologue: stage first adj tile (each thread copies exactly the 16B it will consume)
  async_b128_to_lds(arow + jq, &AdjT[0][ii * 32 + jq]);
#endif

  for (int kb = 0; kb < NN; kb += 32) {
#if ASYNC_LDS
    const int par = (kb >> 5) & 1;
    if (kb + 32 < NN) {
      async_b128_to_lds(arow + (kb + 32) + jq, &AdjT[par ^ 1][ii * 32 + jq]);
      __builtin_amdgcn_s_wait_asynccnt(1);   // current tile resident
    } else {
      __builtin_amdgcn_s_wait_asynccnt(0);
    }
#else
    if (kb + 32 < NN) __builtin_prefetch(arow + (kb + 32) + jq, 0, 3);
#endif
    __syncthreads();   // barrier A: prior step's Pt fragment reads complete

    // ---- produce 32x32 probability tile ----
#if ASYNC_LDS
    const int4 av = *(const int4*)&AdjT[par][ii * 32 + jq];
#else
    const int4 av = *(const int4*)(arow + kb + jq);
#endif
    const int j0 = kb + jq;
    float v0 = (av.x > 0) ? lrelu(f1v + f2b[j0])     : MASK_VAL;
    float v1 = (av.y > 0) ? lrelu(f1v + f2b[j0 + 1]) : MASK_VAL;
    float v2 = (av.z > 0) ? lrelu(f1v + f2b[j0 + 2]) : MASK_VAL;
    float v3 = (av.w > 0) ? lrelu(f1v + f2b[j0 + 3]) : MASK_VAL;
    union { _Float16 h[4]; uint2 u2; } pk;
    pk.h[0] = (_Float16)(__expf(v0 - Mbb[j0])     * iSb[j0]);
    pk.h[1] = (_Float16)(__expf(v1 - Mbb[j0 + 1]) * iSb[j0 + 1]);
    pk.h[2] = (_Float16)(__expf(v2 - Mbb[j0 + 2]) * iSb[j0 + 2]);
    pk.h[3] = (_Float16)(__expf(v3 - Mbb[j0 + 3]) * iSb[j0 + 3]);
    *(uint2*)&Pt[ii * 32 + jq] = pk.u2;

    __syncthreads();   // barrier B: tile visible to all waves

    // ---- consume: two A-frags from LDS, two B-frags from global ----
    Frag16 a0, a1, b0, b1;
#pragma unroll
    for (int d = 0; d < 8; ++d) {
      const int ka = kkA(d, half);
      a0.u[d] = *(const uint32_t*)&Pt[lid * 32 + ka];
      a1.u[d] = *(const uint32_t*)&Pt[(16 + lid) * 32 + ka];
      const int kw = (kb + kkB(d, half)) >> 1;
      b0.u[d] = brow0[kw];
      b1.u[d] = brow1[kw];
    }
    c00 = __builtin_amdgcn_wmma_f32_16x16x32_f16(false, a0.v, false, b0.v, (short)0, c00, false, false);
    c01 = __builtin_amdgcn_wmma_f32_16x16x32_f16(false, a0.v, false, b1.v, (short)0, c01, false, false);
    c10 = __builtin_amdgcn_wmma_f32_16x16x32_f16(false, a1.v, false, b0.v, (short)0, c10, false, false);
    c11 = __builtin_amdgcn_wmma_f32_16x16x32_f16(false, a1.v, false, b1.v, (short)0, c11, false, false);
  }

  // epilogue: leaky-relu, write next-layer h16 or final fp32 output
#pragma unroll
  for (int r = 0; r < 8; ++r) {
    const size_t n0 = (size_t)gi0 + r + 8 * half;  // i-subtile 0
    const size_t n1 = n0 + 16;                     // i-subtile 1
    float r00 = lrelu(c00[r]);
    float r01 = lrelu(c01[r]);
    float r10 = lrelu(c10[r]);
    float r11 = lrelu(c11[r]);
    if (write_f32) {
      f32_out[n0 * FF + o0]      = r00;
      f32_out[n0 * FF + o0 + 16] = r01;
      f32_out[n1 * FF + o0]      = r10;
      f32_out[n1 * FF + o0 + 16] = r11;
    } else {
      h16_out[n0 * FF + o0]      = (_Float16)r00;
      h16_out[n0 * FF + o0 + 16] = (_Float16)r01;
      h16_out[n1 * FF + o0]      = (_Float16)r10;
      h16_out[n1 * FF + o0 + 16] = (_Float16)r11;
    }
  }
}

// ---------------- host-side orchestration ----------------
extern "C" void kernel_launch(void* const* d_in, const int* in_sizes, int n_in,
                              void* d_out, int out_size, void* d_ws, size_t ws_size,
                              hipStream_t stream) {
  const float* x     = (const float*)d_in[0];
  const int*   adj   = (const int*)d_in[1];
  const float* W0    = (const float*)d_in[2];
  const float* Wrest = (const float*)d_in[3];
  const float* Avec  = (const float*)d_in[4];
  float*       out   = (float*)d_out;

  char* ws = (char*)d_ws;
  constexpr size_t SZ_H16   = (size_t)BB * NN * FF * sizeof(_Float16); // 8 MiB
  constexpr size_t SZ_WH16  = SZ_H16;                                  // 8 MiB
  constexpr size_t SZ_W16T  = (size_t)FF * FF * sizeof(_Float16);      // 128 KiB
  constexpr size_t SZ_VEC   = (size_t)BB * NN * sizeof(float);         // 64 KiB
  _Float16* h16   = (_Float16*)(ws);
  _Float16* wh16t = (_Float16*)(ws + SZ_H16);
  _Float16* w16t  = (_Float16*)(ws + SZ_H16 + SZ_WH16);
  float*    f1    = (float*)(ws + SZ_H16 + SZ_WH16 + SZ_W16T);
  float*    f2    = (float*)(ws + SZ_H16 + SZ_WH16 + SZ_W16T + SZ_VEC);
  float*    Mb    = (float*)(ws + SZ_H16 + SZ_WH16 + SZ_W16T + 2 * SZ_VEC);
  float*    iS    = (float*)(ws + SZ_H16 + SZ_WH16 + SZ_W16T + 3 * SZ_VEC);

  const int nelem = BB * NN * FF;
  convert_x_kernel<<<nelem / 256, 256, 0, stream>>>(x, h16, nelem);

  for (int layer = 0; layer < 4; ++layer) {
    const float* W = (layer == 0) ? W0 : (Wrest + (size_t)(layer - 1) * FF * FF);
    convert_w_kernel<<<FF, 256, 0, stream>>>(W, w16t);
    gemm_wh_kernel<<<dim3(BB * NN / 16, 4), 128, 0, stream>>>(h16, w16t, wh16t);
    fvec_kernel<<<dim3(NN / 256, BB), 256, 0, stream>>>(wh16t, Avec + layer * 2 * FF, f1, f2);
    colstats_kernel<<<dim3(NN / 256, BB), 256, 0, stream>>>(adj, f1, f2, Mb, iS);
    attn_kernel<<<dim3(BB * NN / 32), 256, 0, stream>>>(adj, f1, f2, Mb, iS, wh16t,
                                                        h16, out, (layer == 3) ? 1 : 0);
  }
}
